// W8A8B32O32LinearWithScaling_47553877901383
// MI455X (gfx1250) — 8-bit matrix operand order repaired and checked by reading, compiled, not run on hardware
//
#include <hip/hip_runtime.h>
#include <stdint.h>

#define MDIM 8192
#define NDIM 4096
#define KDIM 4096

typedef int v8i __attribute__((ext_vector_type(8)));

__device__ __forceinline__ void async_ld_b64(unsigned lds_off, unsigned voff,
                                             const int8_t* sbase) {
  asm volatile("global_load_async_to_lds_b64 %0, %1, %2"
               :: "v"(lds_off), "v"(voff), "s"(sbase) : "memory");
}
__device__ __forceinline__ void async_ld_b128(unsigned lds_off, unsigned voff,
                                              const int8_t* sbase) {
  asm volatile("global_load_async_to_lds_b128 %0, %1, %2"
               :: "v"(lds_off), "v"(voff), "s"(sbase) : "memory");
}


__global__ __launch_bounds__(256) void w8a8_i32_gemm_kernel(
    const int8_t* __restrict__ X,
    const int8_t* __restrict__ W,
    const int*    __restrict__ bias,
    const float*  __restrict__ out_scale,
    const float*  __restrict__ bias_scale,
    int*          __restrict__ out)
{
  __shared__ alignas(32) int8_t sh[3][32 * 1024];

  const int tid    = threadIdx.x;
  const int lane   = tid & 31;
  const int wave   = tid >> 5;
  const int wave_m = wave >> 2;
  const int wave_n = wave & 3;
  const int blockN = blockIdx.x * 128;
  const int blockM = blockIdx.y * 128;

  const int a_seg  = tid & 3;
  const int a_lane = (tid >> 2) & 31;
  const int a_kh   = (tid >> 7) & 1;
  const int b_half = tid & 1;
  const int b_lane = (tid >> 1) & 31;
  const int b_kh   = (tid >> 6) & 1;
  const int b_ntb  = tid >> 7;

  const int8_t* gAbase = X + (size_t)blockM * KDIM;
  const int8_t* gBbase = W + (size_t)blockN * KDIM;

  unsigned voffA[8], voffB[4];
#pragma unroll
  for (int mt = 0; mt < 8; ++mt)
    voffA[mt] = (unsigned)((mt * 16 + (a_lane & 15)) * KDIM
                           + a_kh * 64 + (a_seg >> 1) * 32 + (a_seg & 1) * 8 + (a_lane >> 4) * 16);
#pragma unroll
  for (int i = 0; i < 4; ++i)
    voffB[i] = (unsigned)(((i * 2 + b_ntb) * 16 + (b_lane & 15)) * KDIM
                          + b_kh * 64 + (b_lane >> 4) * 16 + b_half * 32);

  const unsigned ldsBase = (unsigned)(uintptr_t)&sh[0][0];
  const unsigned lA0 = ldsBase + (unsigned)(a_kh * 1024 + a_lane * 32 + a_seg * 8);
  const unsigned lB0 = ldsBase + 16384u + (unsigned)(b_kh * 1024 + b_lane * 32 + b_half * 16);

  auto issue_stage = [&](int kb, int bufsel) {
    const unsigned lbuf = (unsigned)bufsel * 32768u;
    const int8_t* sa = gAbase + kb;
    const int8_t* sb = gBbase + kb;
#pragma unroll
    for (int mt = 0; mt < 8; ++mt)
      async_ld_b64(lA0 + lbuf + (unsigned)mt * 2048u, voffA[mt], sa);
#pragma unroll
    for (int i = 0; i < 4; ++i) {
      const unsigned nt = (unsigned)(i * 2 + b_ntb);
      async_ld_b128(lB0 + lbuf + nt * 2048u, voffB[i], sb);
    }
  };

  v8i acc[4][2] = {};

  const int T = KDIM / 128;
  issue_stage(0, 0);
  issue_stage(128, 1);

  int bcur = 0, bnext = 2;
  for (int t = 0; t < T; ++t) {
    if (t < T - 1) {
      asm volatile("s_wait_asynccnt 12" ::: "memory");
    } else {
      asm volatile("s_wait_asynccnt 0" ::: "memory");
    }
    __syncthreads();

    if (t + 2 < T) issue_stage((t + 2) * 128, bnext);

    const int8_t* bufA = &sh[bcur][0];
    const int8_t* bufB = &sh[bcur][16 * 1024];
#pragma unroll
    for (int kh = 0; kh < 2; ++kh) {
      v8i af[4], bf[2];
#pragma unroll
      for (int a = 0; a < 4; ++a)
        af[a] = *(const v8i*)(bufA + ((wave_m * 4 + a) * 2 + kh) * 1024 + lane * 32);
#pragma unroll
      for (int b = 0; b < 2; ++b)
        bf[b] = *(const v8i*)(bufB + ((wave_n * 2 + b) * 2 + kh) * 1024 + lane * 32);
#pragma unroll
      for (int a = 0; a < 4; ++a)
#pragma unroll
        for (int b = 0; b < 2; ++b)
          acc[a][b] = __builtin_amdgcn_wmma_i32_16x16x64_iu8(
               true, af[a],  true, bf[b],
              acc[a][b],  false,  false);
    }

    const int bold = bcur;
    bcur  = (bcur == 2) ? 0 : bcur + 1;
    bnext = bold;
  }

  const float os  = out_scale[0];
  const float bsc = bias_scale[0];
#pragma unroll
  for (int a = 0; a < 4; ++a) {
#pragma unroll
    for (int b = 0; b < 2; ++b) {
      const int n  = blockN + wave_n * 32 + b * 16 + (lane & 15);
      const float bb = (float)bias[n] * bsc;
      const int m0 = blockM + wave_m * 64 + a * 16 + ((lane >> 4) << 3);
#pragma unroll
      for (int r = 0; r < 8; ++r) {
        const float v = (float)acc[a][b][r] * os + bb;
        out[(size_t)(m0 + r) * NDIM + n] = __float2int_rn(v);
      }
    }
  }
}

extern "C" void kernel_launch(void* const* d_in, const int* in_sizes, int n_in,
                              void* d_out, int out_size, void* d_ws, size_t ws_size,
                              hipStream_t stream) {
  const int8_t* x    = (const int8_t*)d_in[0];
  const int8_t* w    = (const int8_t*)d_in[1];
  const int*    bias = (const int*)d_in[2];
  const float*  os   = (const float*)d_in[3];
  const float*  bs   = (const float*)d_in[4];
  int* out = (int*)d_out;

  dim3 grid(NDIM / 128, MDIM / 128);
  w8a8_i32_gemm_kernel<<<grid, 256, 0, stream>>>(x, w, bias, os, bs, out);
}
